// EVA02Attention_30863634989392
// MI455X (gfx1250) — compile-verified, matrix operand order checked by reading, not run on hardware
//
#include <hip/hip_runtime.h>
#include <hip/hip_bf16.h>
#include <stdint.h>

typedef __attribute__((ext_vector_type(16))) __bf16 v16bf;
typedef __attribute__((ext_vector_type(8)))  float  v8f;
typedef __attribute__((ext_vector_type(4)))  unsigned int v4u;
typedef __attribute__((ext_vector_type(8)))  int v8i_;
typedef __attribute__((ext_vector_type(4)))  int v4i_;

#define B_ 8
#define N_ 1025
#define C_ 1024
#define H_ 16
#define D_ 64
#define M_ (B_*N_)        // 8200 rows
#define C3_ (3*C_)        // 3072
#define SW 1088           // padded score-row width (17*64 = 34*32)

// LDS layout for attn_kernel (dynamic LDS starts at offset 0)
#define SS_OFF   0u
#define QS_OFF   (32u * SW * 4u)            // 139264
#define KS_OFF   (QS_OFF + 4096u)           // 143360 (two 8KB K buffers)
#define RED_OFF  (KS_OFF + 16384u)          // 159744
#define STAT_OFF (RED_OFF + 512u)           // 160256
#define SMEM_SZ  (STAT_OFF + 128u)          // 160384

union FragB { v16bf v; uint4 q[2]; unsigned short u[16]; };

__device__ __forceinline__ unsigned short bf16_rn(float x) {
    unsigned u = __float_as_uint(x);
    u += 0x7fffu + ((u >> 16) & 1u);
    return (unsigned short)(u >> 16);
}

// A/B fragment (16x32 bf16) from LDS, row-major [row][k] with given stride.
// Lane l: m = l&15, half = l>>4; elements 0..7 -> K = half*8 + 0..7,
// elements 8..15 -> K = half*8 + 16..23  (ISA 7.12.2). Two ds_load_b128.
__device__ __forceinline__ v16bf ld_frag16(const unsigned short* lds, int stride,
                                           int row0, int k0) {
    int lane = threadIdx.x & 31;
    int m = lane & 15, hi = lane >> 4;
    const unsigned short* p = lds + (row0 + m) * stride + k0 + hi * 8;
    FragB f;
    f.q[0] = *(const uint4*)(p);
    f.q[1] = *(const uint4*)(p + 16);
    return f.v;
}

// Same fragment but sourced from fp32 LDS with on-the-fly bf16 conversion.
__device__ __forceinline__ v16bf ld_frag_f32(const float* lds, int stride,
                                             int row0, int k0) {
    int lane = threadIdx.x & 31;
    int m = lane & 15, hi = lane >> 4;
    const float* p = lds + (row0 + m) * stride + k0 + hi * 8;
    FragB f;
#pragma unroll
    for (int i = 0; i < 8; ++i) f.u[i] = bf16_rn(p[i]);
#pragma unroll
    for (int i = 0; i < 8; ++i) f.u[8 + i] = bf16_rn(p[16 + i]);
    return f.v;
}

__device__ __forceinline__ v8f wmma_bf16(v16bf a, v16bf b, v8f c) {
    return __builtin_amdgcn_wmma_f32_16x16x32_bf16(false, a, false, b,
                                                   (short)0, c, false, false);
}

// ---------------------------------------------------------------------------
// TDM: 2D tensor-tile DMA, global (bf16, row-major, row length = stride0
// elements) -> LDS at byte offset lds_off. Rows beyond td1 zero-fill (handles
// the ragged N=1025 tail for free). Issued once per calling wave; tracked
// with TENSORcnt.
// ---------------------------------------------------------------------------
__device__ __forceinline__ void tdm_load_2d_bf16(
    const void* gptr, unsigned lds_off,
    unsigned td0, unsigned td1, unsigned long long stride0,
    unsigned tile0, unsigned tile1)
{
    unsigned long long ga = (unsigned long long)(uintptr_t)gptr;
    v4u g0 = { 1u,                                     // count=1 valid D#
               lds_off,                                // lds_addr (bytes)
               (unsigned)ga,                           // global_addr[31:0]
               ((unsigned)((ga >> 32) & 0x01ffffffu))  // global_addr[56:32]
                   | (2u << 30) };                     // type=2 ("image")
    v8i_ g1 = { (int)(1u << 16),                       // data_size=1 (2 bytes)
                (int)((td0 & 0xffffu) << 16),          // tensor_dim0[15:0]
                (int)((td0 >> 16) | ((td1 & 0xffffu) << 16)),
                (int)((td1 >> 16) | (tile0 << 16)),    // tile_dim0
                (int)(tile1 & 0xffffu),                // tile_dim1 (tile_dim2=0)
                (int)(unsigned)(stride0 & 0xffffffffu),
                (int)(unsigned)((stride0 >> 32) & 0xffffu),
                0 };
    v4i_ g2 = { 0, 0, 0, 0 };
    v4i_ g3 = { 0, 0, 0, 0 };
#if __has_include(<hip/amd_detail/amd_gfx1250_TDM.h>)
    v8i_ g4 = { 0, 0, 0, 0, 0, 0, 0, 0 };
    __builtin_amdgcn_tensor_load_to_lds(g0, g1, g2, g3, g4, 0);
#else
    __builtin_amdgcn_tensor_load_to_lds(g0, g1, g2, g3, 0);
#endif
}

// ---------------------------------------------------------------------------
// Kernel 1: qkv = x @ qkv_w + bias, RoPE on q,k (tokens n>=1), scatter to
// Q/K/V bf16 [B,H,N,D]. Workgroup tile 64x64, 4 waves (2x2 of 32x32), K-step 32.
// ---------------------------------------------------------------------------
__global__ __launch_bounds__(128) void qkv_rope_kernel(
    const float* __restrict__ X, const float* __restrict__ W,
    const float* __restrict__ qb, const float* __restrict__ vb,
    const float* __restrict__ rcos, const float* __restrict__ rsin,
    unsigned short* __restrict__ Q, unsigned short* __restrict__ K,
    unsigned short* __restrict__ V)
{
    __shared__ unsigned short As[64 * 32];   // [m][k] bf16
    __shared__ unsigned short Bs[64 * 32];   // [n][k] bf16 (B transposed)
    __shared__ float          Cs[64 * 64];   // fp32 result tile

    const int t  = threadIdx.x;
    const int w  = t >> 5;
    const int wr = (w >> 1) * 32, wc = (w & 1) * 32;
    const int rm = blockIdx.x * 64;
    const int cn = blockIdx.y * 64;

    v8f acc[2][2] = {};

    for (int k0 = 0; k0 < C_; k0 += 32) {
        // A tile: 64 rows x 32 k (fp32 -> bf16), coalesced float4
#pragma unroll
        for (int i = 0; i < 4; ++i) {
            int q4 = t + i * 128;          // 0..511 float4s
            int r = q4 >> 3, cq = q4 & 7;  // 8 float4 per row
            int grow = rm + r; if (grow >= M_) grow = M_ - 1;
            float4 v4 = ((const float4*)(X + (size_t)grow * C_ + k0))[cq];
            ushort4 u;
            u.x = bf16_rn(v4.x); u.y = bf16_rn(v4.y);
            u.z = bf16_rn(v4.z); u.w = bf16_rn(v4.w);
            *(ushort4*)&As[r * 32 + cq * 4] = u;
        }
        // B tile: 32 k-rows x 64 cols, stored transposed [n][k]
#pragma unroll
        for (int i = 0; i < 4; ++i) {
            int q4 = t + i * 128;
            int kr = q4 >> 4, cq = q4 & 15;   // 16 float4 per k-row
            float4 v4 = ((const float4*)(W + (size_t)(k0 + kr) * C3_ + cn))[cq];
            Bs[(cq * 4 + 0) * 32 + kr] = bf16_rn(v4.x);
            Bs[(cq * 4 + 1) * 32 + kr] = bf16_rn(v4.y);
            Bs[(cq * 4 + 2) * 32 + kr] = bf16_rn(v4.z);
            Bs[(cq * 4 + 3) * 32 + kr] = bf16_rn(v4.w);
        }
        // prefetch next K-step tiles into cache while WMMAs run
        if (k0 + 32 < C_) {
            int r = t >> 3, cq = t & 7;
            int grow = rm + r; if (grow >= M_) grow = M_ - 1;
            __builtin_prefetch(X + (size_t)grow * C_ + (k0 + 32) + cq * 4, 0, 1);
            int kr = t >> 4, cq2 = t & 15;
            __builtin_prefetch(W + (size_t)(k0 + 32 + kr) * C3_ + cn + cq2 * 4, 0, 1);
        }
        __syncthreads();
        v16bf a0 = ld_frag16(As, 32, wr,      0);
        v16bf a1 = ld_frag16(As, 32, wr + 16, 0);
        v16bf b0 = ld_frag16(Bs, 32, wc,      0);
        v16bf b1 = ld_frag16(Bs, 32, wc + 16, 0);
        acc[0][0] = wmma_bf16(a0, b0, acc[0][0]);
        acc[0][1] = wmma_bf16(a0, b1, acc[0][1]);
        acc[1][0] = wmma_bf16(a1, b0, acc[1][0]);
        acc[1][1] = wmma_bf16(a1, b1, acc[1][1]);
        __syncthreads();
    }

    // stage fp32 tile to LDS (D-matrix layout: m = r + 8*hi, n = lane&15)
    {
        int lane = t & 31, hi = lane >> 4, nl = lane & 15;
#pragma unroll
        for (int rt = 0; rt < 2; ++rt)
#pragma unroll
            for (int ct = 0; ct < 2; ++ct)
#pragma unroll
                for (int r = 0; r < 8; ++r)
                    Cs[(wr + rt * 16 + r + 8 * hi) * 64 + wc + ct * 16 + nl] =
                        acc[rt][ct][r];
    }
    __syncthreads();

    // bias + RoPE epilogue, pairwise over columns (d even/odd adjacent in LDS)
#pragma unroll
    for (int i = 0; i < 16; ++i) {
        int p = t + i * 128;               // 0..2047 pairs
        int r = p >> 5, c = (p & 31) * 2;
        int gm = rm + r;
        if (gm >= M_) continue;
        int b = gm / N_, n = gm % N_;
        int gcol = cn + c;
        int which = gcol >> 10, rem = gcol & 1023;
        int h = rem >> 6, d = rem & 63;
        float v0 = Cs[r * 64 + c], v1 = Cs[r * 64 + c + 1];
        if (which == 0)      { v0 += qb[rem]; v1 += qb[rem + 1]; }
        else if (which == 2) { v0 += vb[rem]; v1 += vb[rem + 1]; }
        float o0 = v0, o1 = v1;
        if (which < 2 && n > 0) {
            int ri = (n - 1) * D_ + d;
            float c0 = rcos[ri], s0 = rsin[ri];
            float c1 = rcos[ri + 1], s1 = rsin[ri + 1];
            o0 = v0 * c0 - v1 * s0;
            o1 = v1 * c1 + v0 * s1;
        }
        unsigned short* dst = (which == 0) ? Q : (which == 1) ? K : V;
        size_t off = ((size_t)(b * H_ + h) * N_ + n) * D_ + d;
        unsigned pk = (unsigned)bf16_rn(o0) | ((unsigned)bf16_rn(o1) << 16);
        *(unsigned*)(dst + off) = pk;       // d even -> 4B aligned
    }
}

// ---------------------------------------------------------------------------
// Kernel 2: per (b,h,q-tile of 32): full-row softmax attention.
// Entire 32x1088 fp32 score strip lives in LDS (CDNA5: 320KB/WGP).
// Q tile + K chunks DMA'd by the Tensor Data Mover (double-buffered K ring);
// TENSORcnt waits publish tiles, workgroup barrier distributes them.
// ---------------------------------------------------------------------------
__global__ __launch_bounds__(128) void attn_kernel(
    const unsigned short* __restrict__ Q, const unsigned short* __restrict__ K,
    const unsigned short* __restrict__ V,
    float* __restrict__ attn_out, float* __restrict__ O)
{
    extern __shared__ char smem[];
    float*          Ss = (float*)(smem + SS_OFF);
    unsigned short* Qs = (unsigned short*)(smem + QS_OFF);
    float* red     = (float*)(smem + RED_OFF);
    float* rowstat = (float*)(smem + STAT_OFF);

    const int t  = threadIdx.x;
    const int w  = t >> 5;
    const int q0 = blockIdx.x * 32;
    const int h  = blockIdx.y, b = blockIdx.z;
    const int bh = b * H_ + h;
    const size_t basebh = (size_t)bh * N_ * D_;
    const unsigned short* Qg = Q + basebh;
    const unsigned short* Kg = K + basebh;

    // TDM: Q tile (32 tokens) + first K chunk; rows past the end zero-fill.
    if (w == 0) {
        tdm_load_2d_bf16(Qg + (size_t)q0 * D_, QS_OFF, D_, (unsigned)(N_ - q0),
                         D_, D_, 32);
        tdm_load_2d_bf16(Kg, KS_OFF, D_, (unsigned)N_, D_, D_, 64);
    }

    const float scale = 0.125f;            // D^-0.5
    // --- S = scale * Q K^T, strip into LDS; K chunks double-buffered ---
    for (int i = 0; i < 17; ++i) {
        int kb = i * 64;
        if (w == 0) {
            if (i + 1 < 17) {
                int kb2 = kb + 64;
                tdm_load_2d_bf16(Kg + (size_t)kb2 * D_,
                                 KS_OFF + (unsigned)((i + 1) & 1) * 8192u,
                                 D_, (unsigned)(N_ - kb2), D_, D_, 64);
                __builtin_amdgcn_s_wait_tensorcnt(1); // prev chunk (+Q) done
            } else {
                __builtin_amdgcn_s_wait_tensorcnt(0);
            }
        }
        __syncthreads();
        const unsigned short* Ks =
            (const unsigned short*)(smem + KS_OFF + (unsigned)(i & 1) * 8192u);
        v8f s0 = {}, s1 = {};
#pragma unroll
        for (int ks = 0; ks < 64; ks += 32) {
            v16bf a0 = ld_frag16(Qs, 64, 0,  ks);
            v16bf a1 = ld_frag16(Qs, 64, 16, ks);
            v16bf bb = ld_frag16(Ks, 64, w * 16, ks);
            s0 = wmma_bf16(a0, bb, s0);
            s1 = wmma_bf16(a1, bb, s1);
        }
        int lane = t & 31, hi = lane >> 4, nl = lane & 15;
        int col = kb + w * 16 + nl;
#pragma unroll
        for (int r = 0; r < 8; ++r) {
            Ss[(r + 8 * hi) * SW + col]      = s0[r] * scale;
            Ss[(16 + r + 8 * hi) * SW + col] = s1[r] * scale;
        }
        __syncthreads();
    }
    // kill padded columns (tokens 1025..1087)
    for (int idx = t; idx < 32 * (SW - N_); idx += 128) {
        int r = idx / (SW - N_), c = N_ + idx % (SW - N_);
        Ss[r * SW + c] = -1e30f;
    }
    __syncthreads();
    // row max (4 threads per row)
    {
        int r = t >> 2, qp = t & 3;
        float mx = -1e30f;
        for (int c = qp; c < N_; c += 4) mx = fmaxf(mx, Ss[r * SW + c]);
        red[r * 4 + qp] = mx;
    }
    __syncthreads();
    if (t < 32)
        rowstat[t] = fmaxf(fmaxf(red[t * 4], red[t * 4 + 1]),
                           fmaxf(red[t * 4 + 2], red[t * 4 + 3]));
    __syncthreads();
    // exp + row sum (full padded width: pads become 0)
    {
        int r = t >> 2, qp = t & 3;
        float m = rowstat[r], s = 0.f;
        for (int c = qp; c < SW; c += 4) {
            float p = __expf(Ss[r * SW + c] - m);
            Ss[r * SW + c] = p; s += p;
        }
        red[r * 4 + qp] = s;
    }
    __syncthreads();
    if (t < 32)
        rowstat[t] = 1.0f / (red[t * 4] + red[t * 4 + 1] +
                             red[t * 4 + 2] + red[t * 4 + 3]);
    __syncthreads();
    // normalize in place + write attn
    {
        int r = t >> 2, qp = t & 3;
        float inv = rowstat[r];
        int qr = q0 + r;
        bool valid = qr < N_;
        float* dst = attn_out + ((size_t)bh * N_ + qr) * N_;
        for (int c = qp; c < SW; c += 4) {
            float p = Ss[r * SW + c] * inv;
            Ss[r * SW + c] = p;
            if (valid && c < N_) dst[c] = p;
        }
    }
    __syncthreads();

    // --- O = P @ V, 32 tokens per chunk, V transposed into LDS ---
    unsigned short* Vt = (unsigned short*)(smem + KS_OFF); // [64 d][32 tok]
    v8f o0 = {}, o1 = {};
    for (int kc = 0; kc < SW / 32; ++kc) {
        int tb = kc * 32;
        __syncthreads();
        {
            int tt = t >> 2;
            int tok = tb + tt; if (tok > N_ - 1) tok = N_ - 1;
            int db = (t & 3) * 16;
            const uint4* src = (const uint4*)(V + basebh + (size_t)tok * D_ + db);
            FragB tmp; tmp.q[0] = src[0]; tmp.q[1] = src[1];
#pragma unroll
            for (int i = 0; i < 16; ++i) Vt[(db + i) * 32 + tt] = tmp.u[i];
        }
        __syncthreads();
        v16bf a0 = ld_frag_f32(Ss, SW, 0,  tb);
        v16bf a1 = ld_frag_f32(Ss, SW, 16, tb);
        v16bf bb = ld_frag16(Vt, 32, w * 16, 0);
        o0 = wmma_bf16(a0, bb, o0);
        o1 = wmma_bf16(a1, bb, o1);
    }
    // write O as [B*N, C] fp32 (col = h*64 + d)
    {
        int lane = t & 31, hi = lane >> 4, nl = lane & 15;
        int dcol = h * D_ + w * 16 + nl;
#pragma unroll
        for (int r = 0; r < 8; ++r) {
            int q1 = q0 + r + 8 * hi;
            int q2 = q0 + 16 + r + 8 * hi;
            if (q1 < N_) O[((size_t)(b * N_ + q1)) * C_ + dcol] = o0[r];
            if (q2 < N_) O[((size_t)(b * N_ + q2)) * C_ + dcol] = o1[r];
        }
    }
}

// ---------------------------------------------------------------------------
// Kernel 3: out = O @ proj_w + proj_b (fp32 out), same WMMA tiling as K1.
// ---------------------------------------------------------------------------
__global__ __launch_bounds__(128) void proj_kernel(
    const float* __restrict__ A, const float* __restrict__ W,
    const float* __restrict__ pb, float* __restrict__ out)
{
    __shared__ unsigned short As[64 * 32];
    __shared__ unsigned short Bs[64 * 32];

    const int t  = threadIdx.x;
    const int w  = t >> 5;
    const int wr = (w >> 1) * 32, wc = (w & 1) * 32;
    const int rm = blockIdx.x * 64;
    const int cn = blockIdx.y * 64;

    v8f acc[2][2] = {};

    for (int k0 = 0; k0 < C_; k0 += 32) {
#pragma unroll
        for (int i = 0; i < 4; ++i) {
            int q4 = t + i * 128;
            int r = q4 >> 3, cq = q4 & 7;
            int grow = rm + r; if (grow >= M_) grow = M_ - 1;
            float4 v4 = ((const float4*)(A + (size_t)grow * C_ + k0))[cq];
            ushort4 u;
            u.x = bf16_rn(v4.x); u.y = bf16_rn(v4.y);
            u.z = bf16_rn(v4.z); u.w = bf16_rn(v4.w);
            *(ushort4*)&As[r * 32 + cq * 4] = u;
        }
#pragma unroll
        for (int i = 0; i < 4; ++i) {
            int q4 = t + i * 128;
            int kr = q4 >> 4, cq = q4 & 15;
            float4 v4 = ((const float4*)(W + (size_t)(k0 + kr) * C_ + cn))[cq];
            Bs[(cq * 4 + 0) * 32 + kr] = bf16_rn(v4.x);
            Bs[(cq * 4 + 1) * 32 + kr] = bf16_rn(v4.y);
            Bs[(cq * 4 + 2) * 32 + kr] = bf16_rn(v4.z);
            Bs[(cq * 4 + 3) * 32 + kr] = bf16_rn(v4.w);
        }
        if (k0 + 32 < C_) {
            int r = t >> 3, cq = t & 7;
            int grow = rm + r; if (grow >= M_) grow = M_ - 1;
            __builtin_prefetch(A + (size_t)grow * C_ + (k0 + 32) + cq * 4, 0, 1);
            int kr = t >> 4, cq2 = t & 15;
            __builtin_prefetch(W + (size_t)(k0 + 32 + kr) * C_ + cn + cq2 * 4, 0, 1);
        }
        __syncthreads();
        v16bf a0 = ld_frag16(As, 32, wr,      0);
        v16bf a1 = ld_frag16(As, 32, wr + 16, 0);
        v16bf b0 = ld_frag16(Bs, 32, wc,      0);
        v16bf b1 = ld_frag16(Bs, 32, wc + 16, 0);
        acc[0][0] = wmma_bf16(a0, b0, acc[0][0]);
        acc[0][1] = wmma_bf16(a0, b1, acc[0][1]);
        acc[1][0] = wmma_bf16(a1, b0, acc[1][0]);
        acc[1][1] = wmma_bf16(a1, b1, acc[1][1]);
        __syncthreads();
    }

    int lane = t & 31, hi = lane >> 4, nl = lane & 15;
#pragma unroll
    for (int rt = 0; rt < 2; ++rt)
#pragma unroll
        for (int ct = 0; ct < 2; ++ct) {
            int col = cn + wc + ct * 16 + nl;
            float bias = pb[col];
#pragma unroll
            for (int r = 0; r < 8; ++r) {
                int row = rm + wr + rt * 16 + r + 8 * hi;
                if (row < M_) out[(size_t)row * C_ + col] = acc[rt][ct][r] + bias;
            }
        }
}

// ---------------------------------------------------------------------------
extern "C" void kernel_launch(void* const* d_in, const int* in_sizes, int n_in,
                              void* d_out, int out_size, void* d_ws, size_t ws_size,
                              hipStream_t stream)
{
    const float* x      = (const float*)d_in[0];
    const float* qkv_w  = (const float*)d_in[1];
    const float* q_bias = (const float*)d_in[2];
    const float* v_bias = (const float*)d_in[3];
    const float* proj_w = (const float*)d_in[4];
    const float* proj_b = (const float*)d_in[5];
    const float* rcos   = (const float*)d_in[6];
    const float* rsin   = (const float*)d_in[7];

    float* out  = (float*)d_out;                       // [B,N,C]
    float* attn = out + (size_t)B_ * N_ * C_;          // [B,H,N,N]

    const size_t qkvElems = (size_t)B_ * H_ * N_ * D_; // 8,396,800
    unsigned short* Qbf = (unsigned short*)d_ws;
    unsigned short* Kbf = Qbf + qkvElems;
    unsigned short* Vbf = Kbf + qkvElems;
    float*          Ows = (float*)(Vbf + qkvElems);    // [B*N, C] fp32

    dim3 blk(128);

    dim3 g1((M_ + 63) / 64, C3_ / 64);                 // 129 x 48
    qkv_rope_kernel<<<g1, blk, 0, stream>>>(x, qkv_w, q_bias, v_bias,
                                            rcos, rsin, Qbf, Kbf, Vbf);

    dim3 g2((N_ + 31) / 32, H_, B_);                   // 33 x 16 x 8
    attn_kernel<<<g2, blk, SMEM_SZ, stream>>>(Qbf, Kbf, Vbf, attn, Ows);

    dim3 g3((M_ + 63) / 64, C_ / 64);                  // 129 x 16
    proj_kernel<<<g3, blk, 0, stream>>>(Ows, proj_w, proj_b, out);
}
